// seq2seq_61117384622216
// MI455X (gfx1250) — matrix operand order repaired and checked by reading, not compiled, not run on hardware
//
#include <hip/hip_runtime.h>
#include <hip/hip_bf16.h>

// ---------------------------------------------------------------------------
// Problem constants
// ---------------------------------------------------------------------------
#define BB 64
#define TT 512
#define DD 64
#define HH 256
#define G3 (3 * HH)          // 768 gate columns (r,z,n)
#define MROWS (BB * TT)      // 32768

typedef __attribute__((ext_vector_type(16))) __bf16 v16bf;
typedef __attribute__((ext_vector_type(8)))  float  v8f;

// ---------------------------------------------------------------------------
// WMMA helper: D = A(16x32 bf16) * B(32x16 bf16) + C(16x16 f32)
// ---------------------------------------------------------------------------
__device__ __forceinline__ v8f wmma_bf16(v16bf a, v16bf b, v8f c) {
    return __builtin_amdgcn_wmma_f32_16x16x32_bf16(
        /*neg_a=*/false, a, /*neg_b=*/false, b,
        /*c_mod=*/(short)0, c, /*reuse_a=*/false, /*reuse_b=*/false);
}

// A-fragment K index for element e, lane-half g (ISA 16-bit A 16x32 layout).
__device__ __forceinline__ int a_frag_k(int e, int g) {
    int v = e >> 1, o = e & 1;
    return ((v < 4) ? (g * 8 + v * 2) : (16 + g * 8 + (v - 4) * 2)) + o;
}

// ---------------------------------------------------------------------------
// Pack f32 [N,K] row-major weight into fragment-major bf16 B-operand stream:
//   Bp[((nt*KT + kt)*32 + lane)*16 + e] = W[nt*16 + (lane&15), kt*32 + a_frag_k(e, lane>>4)]
// ---------------------------------------------------------------------------
__global__ void pack_b_frag(const float* __restrict__ W, __bf16* __restrict__ Bp,
                            int N, int K) {
    long idx = (long)blockIdx.x * blockDim.x + threadIdx.x;
    if (idx >= (long)N * K) return;
    const int  KT   = K >> 5;
    const int  e    = (int)(idx & 15);
    const int  lane = (int)((idx >> 4) & 31);
    const long rest = idx >> 9;
    const int  kt   = (int)(rest % KT);
    const int  nt   = (int)(rest / KT);
    const int  k    = kt * 32 + a_frag_k(e, lane >> 4);
    const int  n    = (nt << 4) + (lane & 15);
    Bp[idx] = (__bf16)W[(long)n * K + k];
}

// ---------------------------------------------------------------------------
// Pack f32 [M,K] row-major activations into fragment-major bf16 A-operand.
// ---------------------------------------------------------------------------
__global__ void pack_a_frag(const float* __restrict__ X, __bf16* __restrict__ Ap,
                            int M, int K) {
    long idx = (long)blockIdx.x * blockDim.x + threadIdx.x;
    if (idx >= (long)M * K) return;
    const int  KT   = K >> 5;
    const int  e    = (int)(idx & 15);
    const int  lane = (int)((idx >> 4) & 31);
    const long rest = idx >> 9;
    const int  kt   = (int)(rest % KT);
    const int  mt   = (int)(rest / KT);
    const int  k    = kt * 32 + a_frag_k(e, lane >> 4);
    const int  m    = (mt << 4) + (lane & 15);
    Ap[idx] = (__bf16)X[(long)m * K + k];
}

// ---------------------------------------------------------------------------
// GEMM + bias on pre-packed fragments: Y = A(bf16) * B(bf16) + bias.
// One 16x16 tile per wave; K compile-time -> fully unrolled WMMA chain.
// TOUT=true: GEMM rows are (b*T + t); store time-major as Y[t, b, n]
//            so the sequential scan reads one contiguous block per step.
// ---------------------------------------------------------------------------
template <int K, bool TOUT>
__global__ __launch_bounds__(256)
void gemm_frag_wmma(const v16bf* __restrict__ Ap, const v16bf* __restrict__ Bp,
                    const float* __restrict__ bias, float* __restrict__ Y,
                    int M, int N) {
    constexpr int KT = K / 32;
    const int lane  = threadIdx.x & 31;
    const int wavei = threadIdx.x >> 5;
    const int wpb   = blockDim.x >> 5;
    const long tile = (long)blockIdx.x * wpb + wavei;
    const int  tilesN     = N >> 4;
    const long tilesTotal = (long)(M >> 4) * tilesN;
    if (tile >= tilesTotal) return;                   // wave-uniform exit
    const int tm = (int)(tile / tilesN);
    const int tn = (int)(tile - (long)tm * tilesN);

    const v16bf* a = Ap + (long)tm * KT * 32 + lane;  // one 32B load per frag
    const v16bf* b = Bp + (long)tn * KT * 32 + lane;
    v8f acc = {};
#pragma unroll
    for (int kt = 0; kt < KT; ++kt)
        acc = wmma_bf16(a[kt * 32], b[kt * 32], acc);

    const int g  = lane >> 4;
    const int nl = lane & 15;
    const int n  = (tn << 4) + nl;
    const float bv = bias[n];
#pragma unroll
    for (int i = 0; i < 8; ++i) {
        const long row = (long)(tm << 4) + g * 8 + i; // = b*T + t
        const long yi  = TOUT ? (((row & (TT - 1)) * BB + (row >> 9)) * N + n)
                              : (row * N + n);
        Y[yi] = acc[i] + bv;
    }
}

// ---------------------------------------------------------------------------
// GRU sequential scan.  Grid = B/16 independent WGs (batch-parallel).
// Block = 512 thr = 16 waves; wave w owns hidden column tile j0=16w and its
// r/z/n gate tiles (K = H = 256, 24 WMMA/step).
//  - h state in registers (each thread owns the same 8 elements forever)
//  - bf16 A-staging of h in LDS in fragment-major order (2x ds_load_b128)
//  - r/z Whh fragments register-resident across the t-loop (128 VGPRs);
//    n-gate fragments streamed from L0 each step (opaque index defeats LICM)
//    so nothing spills to scratch.
//  - xproj is time-major [T, B, 768]: each step reads a contiguous block
//  - output written as f32 [B,T,H] and as packed bf16 A-fragments
// ---------------------------------------------------------------------------
__global__ __launch_bounds__(512)
void gru_scan(const float* __restrict__ xprojT,  // [T, B, 768] (if useXproj)
              const v16bf* __restrict__ Bw,      // packed Whh^T fragments
              const float* __restrict__ bih,     // [768] (decoder const input)
              const float* __restrict__ bhh,     // [768]
              const float* __restrict__ h0,      // [B,256] (if useH0)
              float* __restrict__ Yf,            // [B,T,256] f32
              __bf16* __restrict__ Yp,           // packed bf16 A-fragments
              int T_, int useXproj, int useH0) {
    __shared__ v16bf hbp[8 * 32];                // packed h fragments, 8 KB

    const int b0   = blockIdx.x * 16;
    const int lane = threadIdx.x & 31;
    const int wave = threadIdx.x >> 5;
    const int g    = lane >> 4;
    const int j    = (wave << 4) + (lane & 15);  // this thread's hidden column

    // Inverse A-fragment map for column j (k index): -> (kt_j, g_j, e_j)
    const int kt_j = j >> 5;
    const int kk   = j & 31;
    const int blk  = kk >> 3, off = kk & 7;
    const int g_j  = blk & 1;
    const int e_j  = (((blk >> 1) << 2) + (off >> 1)) * 2 + (off & 1);
    __bf16* hbs = (__bf16*)hbp;
    const int haddr = ((kt_j * 32 + g_j * 16) << 4) + e_j;   // + m*16

    float hcur[8];
#pragma unroll
    for (int i = 0; i < 8; ++i) {
        const int m = g * 8 + i;
        float v = useH0 ? h0[(long)(b0 + m) * HH + j] : 0.0f;
        hcur[i] = v;
        hbs[haddr + (m << 4)] = (__bf16)v;
    }
    __syncthreads();

    const v16bf* bR = Bw + ((long)(0 * 16 + wave) * 8) * 32 + lane;
    const v16bf* bZ = Bw + ((long)(1 * 16 + wave) * 8) * 32 + lane;
    const v16bf* bN = Bw + ((long)(2 * 16 + wave) * 8) * 32 + lane;

    for (int t = 0; t < T_; ++t) {
        int nb = 0;
        asm volatile("" : "+s"(nb));             // opaque 0: keep n-gate frag
                                                 // loads inside the loop
        v8f accR = {}, accZ = {}, accN = {};
#pragma unroll
        for (int kt = 0; kt < 8; ++kt) {
            const v16bf a = hbp[kt * 32 + lane]; // ds_load_b128 x2
            accR = wmma_bf16(a, bR[kt * 32], accR);      // register-resident
            accZ = wmma_bf16(a, bZ[kt * 32], accZ);      // register-resident
            accN = wmma_bf16(a, bN[kt * 32 + nb], accN); // streamed from L0
        }

        const int trow = t & 15;
#pragma unroll
        for (int i = 0; i < 8; ++i) {
            const int m = g * 8 + i;
            const int b = b0 + m;
            float ir, iz, inn;
            if (useXproj) {
                const long xrow = ((long)t * BB + b) * G3;   // time-major
                ir  = xprojT[xrow + 0 * HH + j];
                iz  = xprojT[xrow + 1 * HH + j];
                inn = xprojT[xrow + 2 * HH + j];
            } else {                              // decoder: zero input
                ir = bih[j]; iz = bih[HH + j]; inn = bih[2 * HH + j];
            }
            const float hr = accR[i] + bhh[0 * HH + j];
            const float hz = accZ[i] + bhh[1 * HH + j];
            const float hn = accN[i] + bhh[2 * HH + j];
            const float r  = 1.0f / (1.0f + __expf(-(ir + hr)));
            const float z  = 1.0f / (1.0f + __expf(-(iz + hz)));
            const float n  = tanhf(inn + r * hn);
            const float hv = (1.0f - z) * n + z * hcur[i];
            hcur[i] = hv;
            Yf[((long)b * T_ + t) * HH + j] = hv;
            // packed A-fragment position for GEMM row r = b*T + t, col j
            const long mt = (long)b * (T_ >> 4) + (t >> 4);
            Yp[(((mt * 8 + kt_j) * 32 + g_j * 16 + trow) << 4) + e_j] = (__bf16)hv;
        }
        if (useXproj && t + 1 < T_)
            __builtin_prefetch(&xprojT[((long)(t + 1) * BB + b0) * G3 + j], 0, 1);
        __syncthreads();                          // all hbp reads done
#pragma unroll
        for (int i = 0; i < 8; ++i)
            hbs[haddr + ((g * 8 + i) << 4)] = (__bf16)hcur[i];
        __syncthreads();                          // h fully republished
    }
}

// ---------------------------------------------------------------------------
// Gather enc_out[b, seq_len[b]-1] -> encoder_hidden output + decoder h0
// ---------------------------------------------------------------------------
__global__ void gather_last(const float* __restrict__ enc,
                            const int* __restrict__ sl,
                            float* __restrict__ out_hidden,
                            float* __restrict__ last) {
    int idx = blockIdx.x * blockDim.x + threadIdx.x;
    if (idx >= BB * HH) return;
    const int b = idx >> 8;
    const int jj = idx & (HH - 1);
    const int t = sl[b] - 1;
    const float v = enc[((long)b * TT + t) * HH + jj];
    out_hidden[idx] = v;
    last[idx] = v;
}

// ---------------------------------------------------------------------------
// Workspace layout (bytes, 256-aligned)
// ---------------------------------------------------------------------------
static constexpr size_t OFF_WT0   = 0;                        // Wih0 pack [768x64]  bf16
static constexpr size_t OFF_WT1   = OFF_WT0   + 98304;        // Wih1 pack [768x256] bf16
static constexpr size_t OFF_WT2   = OFF_WT1   + 393216;
static constexpr size_t OFF_WHH0  = OFF_WT2   + 393216;       // Whh packs
static constexpr size_t OFF_WHH1  = OFF_WHH0  + 393216;
static constexpr size_t OFF_WHH2  = OFF_WHH1  + 393216;
static constexpr size_t OFF_WHHD  = OFF_WHH2  + 393216;
static constexpr size_t OFF_WOUT  = OFF_WHHD  + 393216;       // Wout pack [64x256] bf16
static constexpr size_t OFF_LAST  = OFF_WOUT  + 32768;        // f32 [64,256]
static constexpr size_t OFF_A0    = OFF_LAST  + 65536;        // input pack [32768x64] bf16
static constexpr size_t OFF_PACKA = OFF_A0    + 4194304;      // act pack ping [32768x256] bf16
static constexpr size_t OFF_PACKB = OFF_PACKA + 16777216;     // act pack pong
static constexpr size_t OFF_YF32  = OFF_PACKB + 16777216;     // f32 [B,T,H] (reused)
static constexpr size_t OFF_XPROJ = OFF_YF32  + 33554432;     // f32 [T,B,768] time-major

extern "C" void kernel_launch(void* const* d_in, const int* in_sizes, int n_in,
                              void* d_out, int out_size, void* d_ws, size_t ws_size,
                              hipStream_t stream) {
    (void)in_sizes; (void)n_in; (void)out_size; (void)ws_size;

    const float* x_in  = (const float*)d_in[0];
    const int*   slen  = (const int*)d_in[1];
    const float* Wih[3] = { (const float*)d_in[2],  (const float*)d_in[6],  (const float*)d_in[10] };
    const float* Whh[3] = { (const float*)d_in[3],  (const float*)d_in[7],  (const float*)d_in[11] };
    const float* bih[3] = { (const float*)d_in[4],  (const float*)d_in[8],  (const float*)d_in[12] };
    const float* bhh[3] = { (const float*)d_in[5],  (const float*)d_in[9],  (const float*)d_in[13] };
    // d_in[14] = W_ihd: multiplied by zeros in the reference -> never used.
    const float* Whhd  = (const float*)d_in[15];
    const float* bihd  = (const float*)d_in[16];
    const float* bhhd  = (const float*)d_in[17];
    const float* Wout  = (const float*)d_in[18];
    const float* bout  = (const float*)d_in[19];

    char* ws = (char*)d_ws;
    __bf16* wtP[3]   = { (__bf16*)(ws + OFF_WT0),  (__bf16*)(ws + OFF_WT1),  (__bf16*)(ws + OFF_WT2) };
    __bf16* whhP[3]  = { (__bf16*)(ws + OFF_WHH0), (__bf16*)(ws + OFF_WHH1), (__bf16*)(ws + OFF_WHH2) };
    __bf16* whhdP    = (__bf16*)(ws + OFF_WHHD);
    __bf16* woutP    = (__bf16*)(ws + OFF_WOUT);
    float*  last     = (float*)(ws + OFF_LAST);
    __bf16* a0P      = (__bf16*)(ws + OFF_A0);
    __bf16* packA    = (__bf16*)(ws + OFF_PACKA);
    __bf16* packB    = (__bf16*)(ws + OFF_PACKB);
    float*  yf32     = (float*)(ws + OFF_YF32);
    float*  xproj    = (float*)(ws + OFF_XPROJ);

    float* out_hidden = (float*)d_out;              // [1,B,H] = 16384 f32
    float* out_dec    = (float*)d_out + BB * HH;    // [B,T,D]

    auto cdiv = [](long a, long b) { return (int)((a + b - 1) / b); };

    // --- 1. pack weights into fragment-major bf16 ---------------------------
    pack_b_frag<<<cdiv((long)G3 * DD, 256), 256, 0, stream>>>(Wih[0], wtP[0], G3, DD);
    pack_b_frag<<<cdiv((long)G3 * HH, 256), 256, 0, stream>>>(Wih[1], wtP[1], G3, HH);
    pack_b_frag<<<cdiv((long)G3 * HH, 256), 256, 0, stream>>>(Wih[2], wtP[2], G3, HH);
    for (int l = 0; l < 3; ++l)
        pack_b_frag<<<cdiv((long)G3 * HH, 256), 256, 0, stream>>>(Whh[l], whhP[l], G3, HH);
    pack_b_frag<<<cdiv((long)G3 * HH, 256), 256, 0, stream>>>(Whhd, whhdP, G3, HH);
    pack_b_frag<<<cdiv((long)DD * HH, 256), 256, 0, stream>>>(Wout, woutP, DD, HH);

    // --- 2. pack layer-0 input activations ----------------------------------
    pack_a_frag<<<cdiv((long)MROWS * DD, 256), 256, 0, stream>>>(x_in, a0P, MROWS, DD);

    const int gemmBlocksXP = cdiv((long)(MROWS / 16) * (G3 / 16), 8);  // 8 waves/block

    // --- 3. encoder layer 0 -------------------------------------------------
    gemm_frag_wmma<DD, true><<<gemmBlocksXP, 256, 0, stream>>>(
        (const v16bf*)a0P, (const v16bf*)wtP[0], bih[0], xproj, MROWS, G3);
    gru_scan<<<BB / 16, 512, 0, stream>>>(xproj, (const v16bf*)whhP[0],
        bih[0], bhh[0], last, yf32, packA, TT, 1, 0);

    // --- 4. encoder layer 1 -------------------------------------------------
    gemm_frag_wmma<HH, true><<<gemmBlocksXP, 256, 0, stream>>>(
        (const v16bf*)packA, (const v16bf*)wtP[1], bih[1], xproj, MROWS, G3);
    gru_scan<<<BB / 16, 512, 0, stream>>>(xproj, (const v16bf*)whhP[1],
        bih[1], bhh[1], last, yf32, packB, TT, 1, 0);

    // --- 5. encoder layer 2 -------------------------------------------------
    gemm_frag_wmma<HH, true><<<gemmBlocksXP, 256, 0, stream>>>(
        (const v16bf*)packB, (const v16bf*)wtP[2], bih[2], xproj, MROWS, G3);
    gru_scan<<<BB / 16, 512, 0, stream>>>(xproj, (const v16bf*)whhP[2],
        bih[2], bhh[2], last, yf32, packA, TT, 1, 0);

    // --- 6. encoder_hidden = enc_out[b, seq_len-1] --------------------------
    gather_last<<<cdiv(BB * HH, 256), 256, 0, stream>>>(yf32, slen, out_hidden, last);

    // --- 7. decoder (zero input -> x_proj = b_ihd const), h0 = last ---------
    gru_scan<<<BB / 16, 512, 0, stream>>>(xproj, (const v16bf*)whhdP,
        bihd, bhhd, last, yf32, packB, TT, 0, 1);

    // --- 8. decoder_output = dec_h @ W_out^T + b_out ------------------------
    const int gemmBlocksOut = cdiv((long)(MROWS / 16) * (DD / 16), 8);
    gemm_frag_wmma<HH, false><<<gemmBlocksOut, 256, 0, stream>>>(
        (const v16bf*)packB, (const v16bf*)woutP, bout, out_dec, MROWS, DD);
}
